// AutoCorrelation_73194832658440
// MI455X (gfx1250) — matrix operand order repaired and checked by reading, not compiled, not run on hardware
//
#include <hip/hip_runtime.h>
#include <math.h>

// ---------------------------------------------------------------------------
// AutoCorrelation (Autoformer) for MI455X / gfx1250, wave32 + WMMA bf16.
// B=8, H=8, L=4096, D=64, C=H*D=512, top_k=8.
//
// mean_value[b,tau] = (1/512) * sum_t <Q_b[(t+tau)%L], K_b[t]>  (512-dim rows)
// computed as banded Gram-matrix diagonals with v_wmma_f32_16x16x32_bf16.
// LDS staged via CDNA5 async global->LDS DMA (ASYNCcnt), WMMA inner loop
// software-pipelined (A-frag prefetch distance 1, B-frags register-resident).
// ---------------------------------------------------------------------------

#define BDIM   8
#define HDIM   8
#define LDIM   4096
#define DDIM   64
#define CDIM   512      // H*D
#define TOPK   8
#define QS     136      // padded LDS row stride in ushorts (272B = 68 banks)

typedef __attribute__((ext_vector_type(16))) __bf16        v16bf;
typedef __attribute__((ext_vector_type(8)))  float         v8f;
typedef __attribute__((ext_vector_type(4))) unsigned int   uint4v;
typedef __attribute__((ext_vector_type(4)))  float         f4;

union FragU { v16bf v; uint4v u[2]; };

__device__ __forceinline__ unsigned short f2bf(float f) {
    unsigned int u = __float_as_uint(f);
    u += 0x7FFFu + ((u >> 16) & 1u);           // round-to-nearest-even
    return (unsigned short)(u >> 16);
}

__device__ __forceinline__ v16bf ld_frag2(const unsigned short* p0,
                                          const unsigned short* p1) {
    FragU f;
    f.u[0] = *(const uint4v*)p0;               // ds_load_b128
    f.u[1] = *(const uint4v*)p1;               // ds_load_b128
    return f.v;
}

// CDNA5 async global->LDS copy, 16B per lane.  VDST = LDS byte offset,
// VADDR = 64-bit global address, GV mode (no SADDR).  Tracked by ASYNCcnt.
__device__ __forceinline__ void async_g2l_b128(unsigned lds_byte_off,
                                               const void* gaddr) {
    asm volatile("global_load_async_to_lds_b128 %0, %1, off"
                 : : "v"(lds_byte_off), "v"(gaddr) : "memory");
}
__device__ __forceinline__ void wait_async0() {
    asm volatile("s_wait_asynccnt 0x0" : : : "memory");
}

// ---------------------------------------------------------------------------
// Kernel 1: pack Q,K f32 (B,H,L,D) -> bf16 (B,L,C) with c = h*64+d
// ---------------------------------------------------------------------------
__global__ __launch_bounds__(256) void pack_bf16_kernel(
    const float* __restrict__ q, const float* __restrict__ k,
    unsigned short* __restrict__ qp, unsigned short* __restrict__ kp)
{
    size_t i = (size_t)blockIdx.x * 256 + threadIdx.x;   // over B*H*L*D = 2^24
    int d = (int)(i & 63);
    size_t r = i >> 6;
    int l = (int)(r & (LDIM - 1)); r >>= 12;
    int h = (int)(r & (HDIM - 1));
    int b = (int)(r >> 3);
    size_t dst = (((size_t)b * LDIM + l) * CDIM) + (size_t)h * DDIM + d;
    qp[dst] = f2bf(q[i]);
    kp[dst] = f2bf(k[i]);
}

// ---------------------------------------------------------------------------
// Kernel 2: banded-Gram correlation.
// grid = (L/128 windows, B).  256 threads = 8 waves.
// Wave w owns j-tile jt=w and i-tiles a = w..w+8; 9 persistent v8f accums.
// tau_local of C element (M,N) of job p = 16p + M - N  (t/c independent!)
// -> scatter to LDS lag accumulator exactly once, at kernel end.
// ---------------------------------------------------------------------------
__global__ __launch_bounds__(256) void corr_kernel(
    const unsigned short* __restrict__ qp,
    const unsigned short* __restrict__ kp,
    float* __restrict__ mv)
{
    extern __shared__ unsigned char smem[];
    unsigned short* ldsQ = (unsigned short*)smem;                       // 256*QS*2 = 69632 B
    unsigned short* ldsK = (unsigned short*)(smem + 256 * QS * 2);      // 128*QS*2 = 34816 B
    float*          ldsR = (float*)(smem + (256 + 128) * QS * 2);       // 128*4    = 512 B

    const int tid  = threadIdx.x;
    const int lane = tid & 31;
    const int w    = tid >> 5;          // wave id == jt
    const int half = lane >> 4;         // lane group
    const int mn   = lane & 15;         // M for A-frag / N for B-frag & C

    const int b    = blockIdx.y;
    const int tau0 = blockIdx.x * 128;

    // LDS byte offsets of the staging buffers (low 32 bits of flat address)
    const unsigned ldsQoff = (unsigned)(unsigned long long)(void*)ldsQ;
    const unsigned ldsKoff = (unsigned)(unsigned long long)(void*)ldsK;

    if (tid < 128) ldsR[tid] = 0.0f;

    v8f acc[9];
#pragma unroll
    for (int p = 0; p < 9; ++p)
        acc[p] = (v8f){0.f, 0.f, 0.f, 0.f, 0.f, 0.f, 0.f, 0.f};

    const size_t base = (size_t)b * LDIM * CDIM;

    for (int t0 = 0; t0 < LDIM; t0 += 128) {
        for (int cch = 0; cch < 4; ++cch) {
            const int c0 = cch * 128;
            __syncthreads();                 // everyone done reading LDS
            // async-DMA Q rows [t0+tau0, t0+tau0+256) mod L, 128 ch (256B/row)
#pragma unroll
            for (int it = 0; it < 16; ++it) {
                int lin = it * 256 + tid;          // 0..4095
                int r  = lin >> 4;
                int cc = lin & 15;
                int l  = (t0 + tau0 + r) & (LDIM - 1);
                const void* g = qp + base + (size_t)l * CDIM + c0 + cc * 8;
                async_g2l_b128(ldsQoff + (unsigned)(r * QS + cc * 8) * 2, g);
            }
            // async-DMA K rows [t0, t0+128)
#pragma unroll
            for (int it = 0; it < 8; ++it) {
                int lin = it * 256 + tid;          // 0..2047
                int r  = lin >> 4;
                int cc = lin & 15;
                int l  = t0 + r;
                const void* g = kp + base + (size_t)l * CDIM + c0 + cc * 8;
                async_g2l_b128(ldsKoff + (unsigned)(r * QS + cc * 8) * 2, g);
            }
            wait_async0();                   // my DMAs landed in LDS
            __syncthreads();                 // everyone's DMAs landed

            // preload all four B fragments (register-resident, 32 VGPRs)
            v16bf bf[4];
#pragma unroll
            for (int kk = 0; kk < 4; ++kk) {
                const unsigned short* bp =
                    ldsK + (16 * w + mn) * QS + kk * 32 + 8 * half;
                bf[kk] = ld_frag2(bp, bp + 16);
            }

            // 36-step pipeline over (kk,p), A-frag prefetch distance 1
            const unsigned short* abase = ldsQ + (16 * w + mn) * QS + 8 * half;
            v16bf a_cur = ld_frag2(abase, abase + 16);   // kk=0, p=0
#pragma unroll
            for (int s = 0; s < 36; ++s) {
                const int kk = s / 9;
                const int p  = s % 9;
                v16bf a_nxt = a_cur;
                if (s < 35) {
                    const int kk1 = (s + 1) / 9;
                    const int p1  = (s + 1) % 9;
                    const unsigned short* ap =
                        ldsQ + (16 * (w + p1) + mn) * QS + kk1 * 32 + 8 * half;
                    a_nxt = ld_frag2(ap, ap + 16);
                }
                acc[p] = __builtin_amdgcn_wmma_f32_16x16x32_bf16(
                    false, a_cur, false, bf[kk], (short)0, acc[p],
                    false, false);
                a_cur = a_nxt;
            }
        }
    }

    // one-time diagonal scatter: element r of acc[p] is (M = r+8*half, N = mn)
#pragma unroll
    for (int p = 0; p < 9; ++p) {
#pragma unroll
        for (int r = 0; r < 8; ++r) {
            int tl = 16 * p + (r + 8 * half) - mn;     // tau - tau0
            if (tl >= 0 && tl < 128)
                atomicAdd(&ldsR[tl], acc[p][r]);
        }
    }
    __syncthreads();
    if (tid < 128)
        mv[(size_t)b * LDIM + tau0 + tid] = ldsR[tid] * (1.0f / (float)CDIM);
}

// ---------------------------------------------------------------------------
// Kernel 3 (single block, 256 thr): batch-mean, top-8 lags, per-b softmax
// ---------------------------------------------------------------------------
__global__ __launch_bounds__(256) void topk_softmax_kernel(
    const float* __restrict__ mv, int* __restrict__ idx_out,
    float* __restrict__ tc_out)
{
    __shared__ float s[LDIM];
    __shared__ float rv[256];
    __shared__ int   ri[256];
    __shared__ int   sidx[TOPK];
    const int tid = threadIdx.x;

    for (int i = tid; i < LDIM; i += 256) {
        float a = 0.f;
        for (int b = 0; b < BDIM; ++b) a += mv[(size_t)b * LDIM + i];
        s[i] = a;
    }
    __syncthreads();

    for (int k = 0; k < TOPK; ++k) {
        float bv = -3.402823466e38f; int bi = 0;
        for (int i = tid; i < LDIM; i += 256)
            if (s[i] > bv) { bv = s[i]; bi = i; }
        rv[tid] = bv; ri[tid] = bi;
        __syncthreads();
        for (int off = 128; off > 0; off >>= 1) {
            if (tid < off && rv[tid + off] > rv[tid]) {
                rv[tid] = rv[tid + off]; ri[tid] = ri[tid + off];
            }
            __syncthreads();
        }
        if (tid == 0) {
            sidx[k] = ri[0];
            idx_out[k] = ri[0];
            s[ri[0]] = -3.402823466e38f;
        }
        __syncthreads();
    }

    if (tid < BDIM) {                    // tid == b
        float wv[TOPK], m = -3.402823466e38f;
        for (int k = 0; k < TOPK; ++k) {
            wv[k] = mv[(size_t)tid * LDIM + sidx[k]];
            m = fmaxf(m, wv[k]);
        }
        float sum = 0.f;
        for (int k = 0; k < TOPK; ++k) { wv[k] = expf(wv[k] - m); sum += wv[k]; }
        float inv = 1.0f / sum;
        for (int k = 0; k < TOPK; ++k) tc_out[tid * TOPK + k] = wv[k] * inv;
    }
}

// ---------------------------------------------------------------------------
// Kernel 4: out[b,h,l,d] = sum_k tc[b,k] * v[b,h,(l+idx[k])%L,d]  (float4)
// ---------------------------------------------------------------------------
__global__ __launch_bounds__(256) void gather_kernel(
    const float* __restrict__ v, const int* __restrict__ idx,
    const float* __restrict__ tc, float* __restrict__ out)
{
    __shared__ int   sidx[TOPK];
    __shared__ float stc[TOPK];

    size_t gid = (size_t)blockIdx.x * 256 + threadIdx.x;   // B*H*L*16 = 2^22
    int d4 = (int)(gid & 15);
    size_t r = gid >> 4;
    int l = (int)(r & (LDIM - 1)); r >>= 12;
    int h = (int)(r & (HDIM - 1));
    int b = (int)(r >> 3);

    if (threadIdx.x < TOPK) {
        sidx[threadIdx.x] = idx[threadIdx.x];
        stc[threadIdx.x]  = tc[b * TOPK + threadIdx.x];    // b uniform in block
    }
    __syncthreads();

    const size_t bh = ((size_t)b * HDIM + h) * LDIM * DDIM;
    f4 acc = (f4){0.f, 0.f, 0.f, 0.f};
#pragma unroll
    for (int k = 0; k < TOPK; ++k) {
        int ls = (l + sidx[k]) & (LDIM - 1);
        f4 val = *(const f4*)(v + bh + (size_t)ls * DDIM + d4 * 4);
        acc += stc[k] * val;
    }
    *(f4*)(out + bh + (size_t)l * DDIM + d4 * 4) = acc;
}

// ---------------------------------------------------------------------------
// Workspace layout (bytes):  qp bf16 32MB | kp bf16 32MB | mv f32 128KB |
//                            idx 32B | tc 256B          (total ~64.2 MB)
// ---------------------------------------------------------------------------
extern "C" void kernel_launch(void* const* d_in, const int* in_sizes, int n_in,
                              void* d_out, int out_size, void* d_ws, size_t ws_size,
                              hipStream_t stream)
{
    const float* q = (const float*)d_in[0];
    const float* k = (const float*)d_in[1];
    const float* v = (const float*)d_in[2];
    float* out = (float*)d_out;

    unsigned char* ws = (unsigned char*)d_ws;
    const size_t QP_OFF  = 0;
    const size_t KP_OFF  = (size_t)BDIM * LDIM * CDIM * 2;          // 32 MB
    const size_t MV_OFF  = KP_OFF + (size_t)BDIM * LDIM * CDIM * 2; // 64 MB
    const size_t IDX_OFF = MV_OFF + (size_t)BDIM * LDIM * 4;
    const size_t TC_OFF  = IDX_OFF + 64;

    unsigned short* qp = (unsigned short*)(ws + QP_OFF);
    unsigned short* kp = (unsigned short*)(ws + KP_OFF);
    float*          mv = (float*)(ws + MV_OFF);
    int*            idx = (int*)(ws + IDX_OFF);
    float*          tc  = (float*)(ws + TC_OFF);

    // 1) pack+transpose to bf16 [B][L][C]
    pack_bf16_kernel<<<(BDIM * HDIM * LDIM * DDIM) / 256, 256, 0, stream>>>(
        q, k, qp, kp);

    // 2) WMMA banded-Gram circular correlation -> mean_value [B][L]
    const size_t smem = (256 + 128) * QS * 2 + 128 * 4;   // 104960 B
    corr_kernel<<<dim3(LDIM / 128, BDIM), 256, smem, stream>>>(qp, kp, mv);

    // 3) top-8 lags + per-batch softmax weights
    topk_softmax_kernel<<<1, 256, 0, stream>>>(mv, idx, tc);

    // 4) weighted circular roll-gather of V -> out [B][H][L][D]
    gather_kernel<<<(BDIM * HDIM * LDIM * (DDIM / 4)) / 256, 256, 0, stream>>>(
        v, idx, tc, out);
}
